// WindowObjectDetector_71897752535063
// MI455X (gfx1250) — matrix operand order repaired and checked by reading, not compiled, not run on hardware
//
#include <hip/hip_runtime.h>
#include <hip/hip_bf16.h>
#include <hip/hip_fp16.h>

// ---------------- problem constants ----------------
#define RR        61
#define CC        61
#define NPATCH    (RR*CC)          // 3721
#define KDIM      12288            // 64*64*3
#define KCHUNKS   (KDIM/32)        // 384
#define NCLS      10
#define NTILES    ((NPATCH+15)/16) // 233
#define IMG_W     1024
#define IMG_ELEMS (1024*1024*3)    // 3145728
#define THRESH_F  0.7f
#define SEGS      (NPATCH+1)       // 3722

typedef __attribute__((ext_vector_type(16))) _Float16     v16h;
typedef __attribute__((ext_vector_type(8)))  _Float16     v8h;
typedef __attribute__((ext_vector_type(8)))  float        v8f;
typedef __attribute__((ext_vector_type(4)))  float        vf4;
typedef __attribute__((ext_vector_type(4)))  unsigned int u32x4;
typedef __attribute__((ext_vector_type(8)))  int          i32x8;
typedef __attribute__((ext_vector_type(4)))  int          i32x4;

// ---------------- kernel 0: image f32 -> f16 (one pass; image then L2-resident) ----
__global__ void img_to_f16(const float* __restrict__ img, _Float16* __restrict__ imgh) {
    int i = (blockIdx.x * blockDim.x + threadIdx.x) * 8;   // IMG_ELEMS divisible by 8
    vf4 a = *(const vf4*)(img + i);
    vf4 b = *(const vf4*)(img + i + 4);
    v8h h;
#pragma unroll
    for (int t = 0; t < 4; ++t) { h[t] = (_Float16)a[t]; h[4 + t] = (_Float16)b[t]; }
    *(v8h*)(imgh + i) = h;
}

// ---------------- kernel 1: pre-pack W into WMMA B fragments via TDM ----------------
// Each block handles one K-chunk (32 rows of W = 1280 contiguous bytes) -> stage the
// chunk into LDS with the Tensor Data Mover, then emit the 32x16 f16 B fragment:
// lanes 0-15 -> column n=lane, K={0..7,16..23}; lanes 16-31 -> K={8..15,24..31} (2 K per VGPR).
__global__ void build_bfrag(const float* __restrict__ W, _Float16* __restrict__ bfrag) {
    __shared__ float wchunk[320];                 // 32 rows x 10 cols
    int kc   = blockIdx.x;
    int lane = threadIdx.x;

    // ---- Tensor DMA descriptor (ISA 8.3/8.4): 2D tensor, 320 x 1 f32 tile ----
    unsigned long long gaddr = (unsigned long long)(const void*)(W + (size_t)kc * 320);
    unsigned int       laddr = (unsigned int)(unsigned long long)(void*)wchunk;

    u32x4 g0;
    g0[0] = 1u;                                            // count=1, is_restore=0
    g0[1] = laddr;                                         // lds_addr (bytes)
    g0[2] = (unsigned int)(gaddr & 0xffffffffu);           // global_addr[31:0]
    g0[3] = ((unsigned int)(gaddr >> 32) & 0x01ffffffu)    // global_addr[56:32]
            | 0x80000000u;                                 // type=2 ("image")
    i32x8 g1;
    g1[0] = (int)(2u << 16);        // workgroup_mask=0, data_size=2 (4 bytes)
    g1[1] = (int)(320u << 16);      // tensor_dim0[15:0]=320 (atomic_barrier_addr=0)
    g1[2] = (int)(1u << 16);        // tensor_dim0 hi=0, tensor_dim1[15:0]=1
    g1[3] = (int)(320u << 16);      // tensor_dim1 hi=0, tile_dim0=320
    g1[4] = 1;                      // tile_dim1=1, tile_dim2=0
    g1[5] = 320;                    // tensor_dim0_stride[31:0]=320
    g1[6] = 0;                      // stride0 hi, tensor_dim1_stride lo
    g1[7] = 0;                      // tensor_dim1_stride hi
    i32x4 gz4 = {};                 // groups 2/3 unused (<=2D tensor)
    i32x8 gz8 = {};                 // extra group (6-arg builtin form): zero

    __builtin_amdgcn_tensor_load_to_lds(g0, g1, gz4, gz4, gz8, 0);
    __builtin_amdgcn_s_wait_tensorcnt(0);
    __syncthreads();

    int n    = lane & 15;                                  // class column (10 -> pad 16)
    int koff = (lane < 16) ? 0 : 8;
    _Float16* dst = bfrag + ((size_t)kc * 32 + lane) * 16;
#pragma unroll
    for (int j = 0; j < 16; ++j) {
        float v = (n < NCLS) ? wchunk[(koff + j + (j & 8)) * NCLS + n] : 0.0f;
        dst[j]  = (_Float16)v;
    }
}

// ---------------- kernel 2: implicit GEMM via v_wmma_f32_16x16x32_f16 ----------------
// One wave per 16-patch tile; 64 patch rows x 6 K-chunks = 384 WMMA steps.
// Per step: 2x global_load_b128 (A halves) + 2x global_load_b128 (B frag) + 1 WMMA.
__global__ void gemm_wmma(const _Float16* __restrict__ imgh,
                          const _Float16* __restrict__ bfrag,
                          const float* __restrict__ bias,
                          float* __restrict__ logits) {
    int tile = blockIdx.x;
    int lane = threadIdx.x;
    int m    = lane & 15;
    int pm   = tile * 16 + m;
    if (pm >= NPATCH) pm = NPATCH - 1;     // clamp addresses; stores guarded later
    int pr = pm / CC, pc = pm % CC;
    const _Float16* pbase = imgh + ((size_t)(pr * 16) * IMG_W + (size_t)(pc * 16)) * 3;
    // A layout: lane<16 holds K {0..7,16..23}; lane>=16 holds K {8..15,24..31}
    int aoff = (lane < 16) ? 0 : 8;

    v8f acc = {};
    for (int y = 0; y < 64; ++y) {
        const _Float16* rowp = pbase + (size_t)y * IMG_W * 3;  // 192 contiguous f16
        if (y < 63) __builtin_prefetch(rowp + IMG_W * 3, 0, 1);
#pragma unroll
        for (int j = 0; j < 6; ++j) {
            const _Float16* ap = rowp + j * 32 + aoff;
            v8h h0 = *(const v8h*)(ap);          // K offsets +0..7
            v8h h1 = *(const v8h*)(ap + 16);     // K offsets +16..23
            v16h af;
#pragma unroll
            for (int t = 0; t < 8; ++t) { af[t] = h0[t]; af[8 + t] = h1[t]; }
            int  kc = y * 6 + j;
            v16h bf = *(const v16h*)(bfrag + ((size_t)kc * 32 + lane) * 16);
            acc = __builtin_amdgcn_wmma_f32_16x16x32_f16(
                    /*neg_a=*/false, af, /*neg_b=*/false, bf,
                    /*c_mod=*/(short)0, acc, /*reuse_a=*/false, /*reuse_b=*/false);
        }
    }
    // epilogue: C layout -> VGPR i holds M=i (lanes 0-15) / M=8+i (lanes 16-31)
    int n     = lane & 15;
    float bv  = (n < NCLS) ? bias[n] : 0.0f;
    int mbase = (lane >> 4) * 8;
    int prow  = tile * 16;
#pragma unroll
    for (int i = 0; i < 8; ++i) {
        int gm = prow + mbase + i;
        if (gm < NPATCH) logits[(size_t)gm * 16 + n] = acc[i] + bv;
    }
}

// ---------------- kernel 3: softmax + threshold seeding ----------------
__global__ void softmax_mask(const float* __restrict__ logits,
                             float* __restrict__ prob_out,
                             int* __restrict__ labelsA) {
    int idx = blockIdx.x * blockDim.x + threadIdx.x;
    if (idx >= NPATCH) return;
    float v[NCLS], mx = -1e30f;
#pragma unroll
    for (int k = 0; k < NCLS; ++k) { v[k] = logits[(size_t)idx * 16 + k]; mx = fmaxf(mx, v[k]); }
    float s = 0.0f;
#pragma unroll
    for (int k = 0; k < NCLS; ++k) { v[k] = __expf(v[k] - mx); s += v[k]; }
    float inv = 1.0f / s;
#pragma unroll
    for (int k = 0; k < NCLS; ++k) {
        float p = v[k] * inv;
        prob_out[(size_t)idx * NCLS + k] = p;
        labelsA[(size_t)k * NPATCH + idx] = (p > THRESH_F) ? (idx + 1) : 0;
    }
}

// ---------------- kernel 4: per-class connected components, fully in LDS ----------------
__global__ void cc_kernel(int* __restrict__ labelsA, float* __restrict__ labels_out) {
    __shared__ int cur[NPATCH];
    __shared__ int changed;
    int k = blockIdx.x;
    int t = threadIdx.x;
    int* src = labelsA + (size_t)k * NPATCH;
    for (int i = t; i < NPATCH; i += 1024) cur[i] = src[i];
    __syncthreads();
    for (;;) {
        if (t == 0) changed = 0;
        __syncthreads();
        int nv[4], id[4], cnt = 0;
        for (int i = t; i < NPATCH; i += 1024) {            // Jacobi read phase
            int self = cur[i], best = self;
            if (self) {
                int r = i / CC, c = i % CC;
                if (r > 0)      { int o = cur[i - CC]; if (o && o < best) best = o; }
                if (r < RR - 1) { int o = cur[i + CC]; if (o && o < best) best = o; }
                if (c > 0)      { int o = cur[i - 1];  if (o && o < best) best = o; }
                if (c < CC - 1) { int o = cur[i + 1];  if (o && o < best) best = o; }
            }
            nv[cnt] = best; id[cnt] = i; ++cnt;
        }
        __syncthreads();
        for (int q = 0; q < cnt; ++q)                        // write phase
            if (nv[q] != cur[id[q]]) { cur[id[q]] = nv[q]; changed = 1; }
        __syncthreads();
        if (!changed) break;
    }
    for (int i = t; i < NPATCH; i += 1024) {
        src[i] = cur[i];                                     // keep for boxes pass
        labels_out[(size_t)i * NCLS + k] = (float)cur[i];    // (R,C,10) flattened
    }
}

// ---------------- kernels 5-7: bounding boxes ----------------
__global__ void boxes_init(int* rmin, int* rmax, int* cmin, int* cmax, int* cnt) {
    int i = blockIdx.x * blockDim.x + threadIdx.x;
    if (i >= NCLS * SEGS) return;
    rmin[i] =  0x7fffffff; cmin[i] =  0x7fffffff;
    rmax[i] = -0x7fffffff; cmax[i] = -0x7fffffff;
    cnt[i]  = 0;
}

__global__ void boxes_acc(const int* __restrict__ labelsA,
                          int* rmin, int* rmax, int* cmin, int* cmax, int* cnt) {
    int idx = blockIdx.x * blockDim.x + threadIdx.x;
    if (idx >= NCLS * NPATCH) return;
    int k = idx / NPATCH, i = idx % NPATCH;
    int L = labelsA[idx];            // 0..NPATCH; 0 = background segment (masked later)
    int r = i / CC, c = i % CC;
    int s = k * SEGS + L;
    atomicMin(&rmin[s], r); atomicMax(&rmax[s], r);
    atomicMin(&cmin[s], c); atomicMax(&cmax[s], c);
    atomicAdd(&cnt[s], 1);
}

__global__ void boxes_fin(const int* rmin, const int* rmax, const int* cmin, const int* cmax,
                          const int* cnt, float* __restrict__ boxes_out,
                          float* __restrict__ valid_out) {
    int i = blockIdx.x * blockDim.x + threadIdx.x;
    if (i >= NCLS * SEGS) return;
    int s = i % SEGS;
    bool valid = (cnt[i] > 0) && (s > 0);
    float* bo = boxes_out + (size_t)i * 4;
    if (valid) {
        bo[0] = (float)(rmin[i] - 1); bo[1] = (float)(cmin[i] - 1);
        bo[2] = (float)(rmax[i] + 1); bo[3] = (float)(cmax[i] + 1);
    } else {
        bo[0] = bo[1] = bo[2] = bo[3] = -1.0f;
    }
    valid_out[i] = valid ? 1.0f : 0.0f;
}

// ---------------- launcher ----------------
extern "C" void kernel_launch(void* const* d_in, const int* in_sizes, int n_in,
                              void* d_out, int out_size, void* d_ws, size_t ws_size,
                              hipStream_t stream) {
    const float* img  = (const float*)d_in[0];   // (1,1024,1024,3) f32
    const float* W    = (const float*)d_in[1];   // (12288,10) f32
    const float* bias = (const float*)d_in[2];   // (10,) f32

    float* out        = (float*)d_out;
    float* prob_out   = out;                         // 37210
    float* labels_out = out + 37210;                 // 37210
    float* boxes_out  = out + 74420;                 // 148880
    float* valid_out  = out + 223300;                // 37220

    int* ws       = (int*)d_ws;
    float* logits = (float*)ws;                      // 3721*16 = 59536 elems
    int* labelsA  = ws + 59536;                      // 10*3721 = 37210
    int* rmin     = ws + 96746;                      // 10*3722 = 37220 each
    int* rmax     = ws + 133966;
    int* cmin     = ws + 171186;
    int* cmax     = ws + 208406;
    int* cnt      = ws + 245626;                     // ends at elem 282846
    _Float16* bfrag = (_Float16*)((char*)d_ws + (size_t)282880 * 4);      // 384 KB
    _Float16* imgh  = (_Float16*)((char*)d_ws + (size_t)282880 * 4 + 393216); // 6 MB

    img_to_f16  <<<IMG_ELEMS / (8 * 256), 256, 0, stream>>>(img, imgh);
    build_bfrag <<<KCHUNKS, 32, 0, stream>>>(W, bfrag);
    gemm_wmma   <<<NTILES, 32, 0, stream>>>(imgh, bfrag, bias, logits);
    softmax_mask<<<(NPATCH + 255) / 256, 256, 0, stream>>>(logits, prob_out, labelsA);
    cc_kernel   <<<NCLS, 1024, 0, stream>>>(labelsA, labels_out);
    boxes_init  <<<(NCLS * SEGS + 255) / 256, 256, 0, stream>>>(rmin, rmax, cmin, cmax, cnt);
    boxes_acc   <<<(NCLS * NPATCH + 255) / 256, 256, 0, stream>>>(labelsA, rmin, rmax, cmin, cmax, cnt);
    boxes_fin   <<<(NCLS * SEGS + 255) / 256, 256, 0, stream>>>(rmin, rmax, cmin, cmax, cnt, boxes_out, valid_out);
}
